// ReGCN_43361989820515
// MI455X (gfx1250) — matrix operand order repaired and checked by reading, not compiled, not run on hardware
//
#include <hip/hip_runtime.h>
#include <math.h>

// ---------------------------------------------------------------------------
// ReGCN pipeline for MI455X (gfx1250, wave32).
//  - Small 128-K node GEMMs: exact fp32 V_WMMA_F32_16X16X4_F32.
//  - Big 1024-K adj/attn GEMMs: split-f16 (hi+lo) emulated-fp32 via
//    3x V_WMMA_F32_16X16X32_F16 per K=32 step (drops only the lo*lo term,
//    ~2^-22 relative) -> ~2.7x the fp32-WMMA matrix throughput, which lands
//    at the HBM/L2 roofline for the 64MB adj matrix.
// ---------------------------------------------------------------------------

typedef __attribute__((ext_vector_type(2)))  float    v2f;
typedef __attribute__((ext_vector_type(8)))  float    v8f;
typedef __attribute__((ext_vector_type(8)))  _Float16 v8h;
typedef __attribute__((ext_vector_type(16))) _Float16 v16h;

#define NTOK 1024
#define NB   16
#define HIDD 128
#define BN   (NB * NTOK)
#define NEGV (-9000000000000000.0f)
#define ALPHAV 0.2f

__device__ __forceinline__ v8f wmma_f32(v2f a, v2f b, v8f c) {
  return __builtin_amdgcn_wmma_f32_16x16x4_f32(false, a, false, b, (short)0, c,
                                               false, false);
}
__device__ __forceinline__ v8f wmma_f16(v16h a, v16h b, v8f c) {
  return __builtin_amdgcn_wmma_f32_16x16x32_f16(false, a, false, b, (short)0, c,
                                                false, false);
}
__device__ __forceinline__ v16h cat8(v8h a, v8h b) {
  return __builtin_shufflevector(a, b, 0, 1, 2, 3, 4, 5, 6, 7,
                                 8, 9, 10, 11, 12, 13, 14, 15);
}

// LDS A-tile layout: element (r, j) at [(j>>3)*128 + r*8 + (j&7)]
// -> a lane's 8-consecutive-K run is one 16B-aligned ds_load_b128 on banks
//    4r..4r+3: conflict-free across the 16 M-lanes, no padding, 32KB/tile.
__device__ __forceinline__ int lds_idx(int r, int j) {
  return ((j >> 3) << 7) + (r << 3) + (j & 7);
}

// ---------------------------------------------------------------------------
// Y[row,col] = (opt relu)(X[row,:] @ W[:,col] + opt bias) — exact fp32 WMMA.
// Block 256 = 8 waves; block covers 16 rows x 128 cols.
// ---------------------------------------------------------------------------
__global__ void node_gemm(const float* __restrict__ X, const float* __restrict__ W,
                          const float* __restrict__ bias, float* __restrict__ Y,
                          int do_relu) {
  const int tid  = threadIdx.x;
  const int wave = tid >> 5;
  const int lane = tid & 31;
  const int half = lane >> 4;
  const int lp   = lane & 15;
  const int row0 = blockIdx.x * 16;
  const int col0 = wave * 16;

  v8f acc = {};
  for (int k0 = 0; k0 < HIDD; k0 += 4) {
    const int kk = k0 + 2 * half;
    v2f a, b;
    a.x = X[(size_t)(row0 + lp) * HIDD + kk];
    a.y = X[(size_t)(row0 + lp) * HIDD + kk + 1];
    b.x = W[(size_t)kk * HIDD + col0 + lp];
    b.y = W[(size_t)(kk + 1) * HIDD + col0 + lp];
    acc = wmma_f32(a, b, acc);
  }

  const int col = col0 + lp;
  for (int v = 0; v < 8; ++v) {
    const int row = row0 + v + 8 * half;
    float val = acc[v];
    if (bias) val += bias[col];
    if (do_relu) val = fmaxf(val, 0.f);
    Y[(size_t)row * HIDD + col] = val;
  }
}

// ---------------------------------------------------------------------------
// Transpose + hi/lo f16 split: X (B,N,128) f32 -> Xhi/Xlo (B,128,1024) f16.
// Column-major output makes B-fragments contiguous along K.
// ---------------------------------------------------------------------------
__global__ void split_transpose(const float* __restrict__ X,
                                _Float16* __restrict__ hi,
                                _Float16* __restrict__ lo) {
  __shared__ float t[32][33];
  const int b  = blockIdx.z;
  const int j0 = blockIdx.x * 32;
  const int h0 = blockIdx.y * 32;
  const int lx = threadIdx.x;   // 32
  const int ly = threadIdx.y;   // 8

  for (int r = 0; r < 4; ++r) {
    const int j = j0 + ly + r * 8;
    t[ly + r * 8][lx] = X[((size_t)(b << 10) + j) * HIDD + h0 + lx];
  }
  __syncthreads();
  for (int r = 0; r < 4; ++r) {
    const int h = h0 + ly + r * 8;
    const int j = j0 + lx;
    const float v = t[lx][ly + r * 8];
    const _Float16 vh = (_Float16)v;
    const size_t o = ((size_t)b * HIDD + h) * NTOK + j;
    hi[o] = vh;
    lo[o] = (_Float16)(v - (float)vh);
  }
}

// ---------------------------------------------------------------------------
// Y[b,i,:] = relu( sum_j adj[b,i,j] * X[b,j,:] ) * mask[b,i] (+ residual)
// adj tile (16x1024) split hi/lo f16 into LDS during staging; split-f16 WMMA.
// Grid: (64, B). Block 256 = 8 waves covering 16 rows x 128 cols.
// ---------------------------------------------------------------------------
__global__ void adj_gemm(const float* __restrict__ adj,
                         const _Float16* __restrict__ Xhi,
                         const _Float16* __restrict__ Xlo,
                         const float* __restrict__ residual,
                         const float* __restrict__ mask, float* __restrict__ Y) {
  __shared__ _Float16 ahi[16 * NTOK];   // 32 KB
  __shared__ _Float16 alo[16 * NTOK];   // 32 KB
  const int tid  = threadIdx.x;
  const int b    = blockIdx.y;
  const int row0 = blockIdx.x * 16;

  const size_t adj_base = ((size_t)b * NTOK + row0) * NTOK;
  for (int idx = tid; idx < 16 * NTOK; idx += 256) {
    const int r = idx >> 10;
    const int j = idx & (NTOK - 1);
    const float v = adj[adj_base + (size_t)r * NTOK + j];
    const _Float16 vh = (_Float16)v;
    const int o = lds_idx(r, j);
    ahi[o] = vh;
    alo[o] = (_Float16)(v - (float)vh);
  }
  __syncthreads();

  const int wave = tid >> 5;
  const int lane = tid & 31;
  const int half = lane >> 4;
  const int lp   = lane & 15;
  const int col0 = wave * 16;
  const _Float16* bhi = Xhi + ((size_t)b * HIDD + col0 + lp) * NTOK;
  const _Float16* blo = Xlo + ((size_t)b * HIDD + col0 + lp) * NTOK;

  v8f acc = {};
  for (int k0 = 0; k0 < NTOK; k0 += 32) {
    const int ka = k0 + 8 * half;    // A: runs [ka,ka+8) and [ka+16,ka+24)
    const int kb = k0 + 8 * half;    // B: runs [kb,kb+8) and [kb+16,kb+24)
    const v16h Ahi = cat8(*(const v8h*)(ahi + lds_idx(lp, ka)),
                          *(const v8h*)(ahi + lds_idx(lp, ka + 16)));
    const v16h Alo = cat8(*(const v8h*)(alo + lds_idx(lp, ka)),
                          *(const v8h*)(alo + lds_idx(lp, ka + 16)));
    const v16h Bhi = cat8(*(const v8h*)(bhi + kb), *(const v8h*)(bhi + kb + 16));
    const v16h Blo = cat8(*(const v8h*)(blo + kb), *(const v8h*)(blo + kb + 16));
    acc = wmma_f16(Ahi, Bhi, acc);
    acc = wmma_f16(Alo, Bhi, acc);
    acc = wmma_f16(Ahi, Blo, acc);
  }

  const int col = col0 + lp;
  for (int v = 0; v < 8; ++v) {
    const int row = row0 + v + 8 * half;
    const size_t node = (size_t)b * NTOK + row;
    float val = fmaxf(acc[v], 0.f) * mask[node];
    if (residual) val += residual[node * HIDD + col];
    Y[node * HIDD + col] = val;
  }
}

// ---------------------------------------------------------------------------
// e1[n] = Wh[n,:].a[0:128] ; e2[n] = Wh[n,:].a[128:256]
// ---------------------------------------------------------------------------
__global__ void gat_scores(const float* __restrict__ Wh, const float* __restrict__ a,
                           float* __restrict__ e1, float* __restrict__ e2) {
  const int n = blockIdx.x * blockDim.x + threadIdx.x;
  const float* row = Wh + (size_t)n * HIDD;
  float s1 = 0.f, s2 = 0.f;
  for (int h = 0; h < HIDD; ++h) {
    const float v = row[h];
    s1 += v * a[h];
    s2 += v * a[HIDD + h];
  }
  e1[n] = s1;
  e2[n] = s2;
}

// ---------------------------------------------------------------------------
// Per row i: m = max_j s_ij, Z = sum_j exp(s_ij - m); literal NEG semantics
// (all-masked row -> uniform softmax, matching jnp.where + softmax).
// ---------------------------------------------------------------------------
__global__ void gat_softmax_stats(const float* __restrict__ adj,
                                  const float* __restrict__ mask,
                                  const float* __restrict__ e1,
                                  const float* __restrict__ e2,
                                  float* __restrict__ mrow, float* __restrict__ zrow) {
  __shared__ float red[256];
  const int bn  = blockIdx.x;
  const int b   = bn >> 10;
  const int i   = bn & (NTOK - 1);
  const int tid = threadIdx.x;
  const float mi  = mask[bn];
  const float e1i = e1[bn];
  const size_t abase = ((size_t)b * NTOK + i) * NTOK;

  float sv[4];
  float lmax = -INFINITY;
  for (int t = 0; t < 4; ++t) {
    const int j = tid + t * 256;
    const float av = adj[abase + j];
    const float mj = mask[(size_t)b * NTOK + j];
    float e = e1i + e2[(size_t)b * NTOK + j];
    e = (e > 0.f) ? e : ALPHAV * e;
    const float s = (av > 0.f && mj > 0.f && mi > 0.f) ? e : NEGV;
    sv[t] = s;
    lmax = fmaxf(lmax, s);
  }
  red[tid] = lmax;
  __syncthreads();
  for (int s = 128; s > 0; s >>= 1) {
    if (tid < s) red[tid] = fmaxf(red[tid], red[tid + s]);
    __syncthreads();
  }
  const float m = red[0];
  __syncthreads();

  float lsum = 0.f;
  for (int t = 0; t < 4; ++t) lsum += __expf(sv[t] - m);
  red[tid] = lsum;
  __syncthreads();
  for (int s = 128; s > 0; s >>= 1) {
    if (tid < s) red[tid] += red[tid + s];
    __syncthreads();
  }
  if (tid == 0) { mrow[bn] = m; zrow[bn] = red[0]; }
}

// ---------------------------------------------------------------------------
// Y[b,i,:] = relu( sum_j attn[b,i,j] * Wh[b,j,:] ) * mask[b,i]
// attn computed + hi/lo-split directly into LDS during staging; then the
// same split-f16 WMMA loop as adj_gemm consumes it.
// ---------------------------------------------------------------------------
__global__ void gat_gemm(const float* __restrict__ adj,
                         const _Float16* __restrict__ Whhi,
                         const _Float16* __restrict__ Whlo,
                         const float* __restrict__ mask, const float* __restrict__ e1,
                         const float* __restrict__ e2, const float* __restrict__ mrow,
                         const float* __restrict__ zrow, float* __restrict__ Y) {
  __shared__ _Float16 ahi[16 * NTOK];
  __shared__ _Float16 alo[16 * NTOK];
  const int tid  = threadIdx.x;
  const int b    = blockIdx.y;
  const int row0 = blockIdx.x * 16;
  const size_t adj_base = ((size_t)b * NTOK + row0) * NTOK;

  for (int idx = tid; idx < 16 * NTOK; idx += 256) {
    const int r = idx >> 10;
    const int j = idx & (NTOK - 1);
    const size_t ni = (size_t)b * NTOK + row0 + r;
    const size_t nj = (size_t)b * NTOK + j;
    const float av = adj[adj_base + (size_t)r * NTOK + j];
    float e = e1[ni] + e2[nj];
    e = (e > 0.f) ? e : ALPHAV * e;
    const float s = (av > 0.f && mask[nj] > 0.f && mask[ni] > 0.f) ? e : NEGV;
    const float attn = __expf(s - mrow[ni]) / zrow[ni];
    const _Float16 vh = (_Float16)attn;
    const int o = lds_idx(r, j);
    ahi[o] = vh;
    alo[o] = (_Float16)(attn - (float)vh);
  }
  __syncthreads();

  const int wave = tid >> 5;
  const int lane = tid & 31;
  const int half = lane >> 4;
  const int lp   = lane & 15;
  const int col0 = wave * 16;
  const _Float16* bhi = Whhi + ((size_t)b * HIDD + col0 + lp) * NTOK;
  const _Float16* blo = Whlo + ((size_t)b * HIDD + col0 + lp) * NTOK;

  v8f acc = {};
  for (int k0 = 0; k0 < NTOK; k0 += 32) {
    const int ka = k0 + 8 * half;
    const int kb = k0 + 8 * half;
    const v16h Ahi = cat8(*(const v8h*)(ahi + lds_idx(lp, ka)),
                          *(const v8h*)(ahi + lds_idx(lp, ka + 16)));
    const v16h Alo = cat8(*(const v8h*)(alo + lds_idx(lp, ka)),
                          *(const v8h*)(alo + lds_idx(lp, ka + 16)));
    const v16h Bhi = cat8(*(const v8h*)(bhi + kb), *(const v8h*)(bhi + kb + 16));
    const v16h Blo = cat8(*(const v8h*)(blo + kb), *(const v8h*)(blo + kb + 16));
    acc = wmma_f16(Ahi, Bhi, acc);
    acc = wmma_f16(Alo, Bhi, acc);
    acc = wmma_f16(Ahi, Blo, acc);
  }

  const int col = col0 + lp;
  for (int v = 0; v < 8; ++v) {
    const int row = row0 + v + 8 * half;
    const size_t node = (size_t)b * NTOK + row;
    Y[node * HIDD + col] = fmaxf(acc[v], 0.f) * mask[node];
  }
}

// ---------------------------------------------------------------------------
// Z[bn,h] = sigmoid(G[bn,:].w_sa + b_sa) * Yln[bn,h] * mask[bn]
// ---------------------------------------------------------------------------
__global__ void soft_att_kernel(const float* __restrict__ G,
                                const float* __restrict__ Yln,
                                const float* __restrict__ w_sa,
                                const float* __restrict__ b_sa,
                                const float* __restrict__ mask,
                                float* __restrict__ Z) {
  __shared__ float red[128];
  const int bn = blockIdx.x;
  const int h  = threadIdx.x;
  red[h] = G[(size_t)bn * HIDD + h] * w_sa[h];
  __syncthreads();
  for (int s = 64; s > 0; s >>= 1) {
    if (h < s) red[h] += red[h + s];
    __syncthreads();
  }
  const float satt = 1.f / (1.f + __expf(-(red[0] + b_sa[0])));
  Z[(size_t)bn * HIDD + h] = satt * Yln[(size_t)bn * HIDD + h] * mask[bn];
}

// ---------------------------------------------------------------------------
// out[b,h] = (sum_n Z[b,n,h]) * (max_n Z[b,n,h])
// ---------------------------------------------------------------------------
__global__ void reduce_kernel(const float* __restrict__ Z, float* __restrict__ out) {
  const int b = blockIdx.x;
  const int h = threadIdx.x;
  float s = 0.f, mx = -INFINITY;
  for (int n = 0; n < NTOK; ++n) {
    const float v = Z[((size_t)b * NTOK + n) * HIDD + h];
    s += v;
    mx = fmaxf(mx, v);
  }
  out[(size_t)b * HIDD + h] = s * mx;
}

// ---------------------------------------------------------------------------
extern "C" void kernel_launch(void* const* d_in, const int* in_sizes, int n_in,
                              void* d_out, int out_size, void* d_ws, size_t ws_size,
                              hipStream_t stream) {
  const float* inputs = (const float*)d_in[0];
  const float* adj    = (const float*)d_in[1];
  const float* mask   = (const float*)d_in[2];
  const float* W0     = (const float*)d_in[3];
  const float* W1     = (const float*)d_in[4];
  const float* Wg     = (const float*)d_in[5];
  const float* a      = (const float*)d_in[6];
  const float* w_sa   = (const float*)d_in[7];
  const float* b_sa   = (const float*)d_in[8];
  const float* W_ln   = (const float*)d_in[9];
  const float* b_ln   = (const float*)d_in[10];
  float* out = (float*)d_out;

  float* ws = (float*)d_ws;
  const size_t BIG = (size_t)BN * HIDD;              // 2M floats
  float* bufA = ws;
  float* bufB = ws + BIG;
  float* bufC = ws + 2 * BIG;
  _Float16* hi = (_Float16*)(ws + 3 * BIG);          // 2M f16 (= BIG/2 floats)
  _Float16* lo = (_Float16*)(ws + 3 * BIG + BIG / 2);
  float* e1   = ws + 4 * BIG;
  float* e2   = e1 + BN;
  float* mrow = e2 + BN;
  float* zrow = mrow + BN;

  const dim3 blk(256);
  const dim3 adj_grid(NTOK / 16, NB);
  const dim3 tr_grid(NTOK / 32, HIDD / 32, NB);
  const dim3 tr_blk(32, 8);
  const int  node_blocks = BN / 16;                  // 1024

  // 1. X0 = inputs @ W0
  node_gemm<<<node_blocks, blk, 0, stream>>>(inputs, W0, nullptr, bufA, 0);
  // 2. split X0 -> hi/lo (column-major)
  split_transpose<<<tr_grid, tr_blk, 0, stream>>>(bufA, hi, lo);
  // 3. H1 = relu(adj @ X0) * mask
  adj_gemm<<<adj_grid, blk, 0, stream>>>(adj, hi, lo, nullptr, mask, bufB);
  // 4. X1 = H1 @ W1
  node_gemm<<<node_blocks, blk, 0, stream>>>(bufB, W1, nullptr, bufA, 0);
  // 5. split X1
  split_transpose<<<tr_grid, tr_blk, 0, stream>>>(bufA, hi, lo);
  // 6. H2 = H1 + relu(adj @ X1) * mask
  adj_gemm<<<adj_grid, blk, 0, stream>>>(adj, hi, lo, bufB, mask, bufC);
  // 7. Wh = H2 @ Wg
  node_gemm<<<node_blocks, blk, 0, stream>>>(bufC, Wg, nullptr, bufA, 0);
  // 8. attention score vectors
  gat_scores<<<BN / 256, blk, 0, stream>>>(bufA, a, e1, e2);
  // 9. per-row softmax stats
  gat_softmax_stats<<<BN, blk, 0, stream>>>(adj, mask, e1, e2, mrow, zrow);
  // 10. split Wh
  split_transpose<<<tr_grid, tr_blk, 0, stream>>>(bufA, hi, lo);
  // 11. G = relu(attn @ Wh) * mask
  gat_gemm<<<adj_grid, blk, 0, stream>>>(adj, hi, lo, mask, e1, e2, mrow, zrow,
                                         bufB);
  // 12. Yln = relu(G @ W_ln + b_ln)
  node_gemm<<<node_blocks, blk, 0, stream>>>(bufB, W_ln, b_ln, bufC, 1);
  // 13. Z = sigmoid(G.w_sa + b_sa) * Yln * mask
  soft_att_kernel<<<BN, dim3(128), 0, stream>>>(bufB, bufC, w_sa, b_sa, mask,
                                                bufA);
  // 14. out = sum_n * max_n
  reduce_kernel<<<NB, dim3(128), 0, stream>>>(bufA, out);
}
